// IntMultiPrecConv2d_23536420782679
// MI455X (gfx1250) — matrix operand order repaired and checked by reading, not compiled, not run on hardware
//
#include <hip/hip_runtime.h>

typedef __attribute__((ext_vector_type(16))) _Float16 v16h;
typedef __attribute__((ext_vector_type(8)))  _Float16 v8h;
typedef __attribute__((ext_vector_type(8)))  float    v8f;
typedef __attribute__((ext_vector_type(4)))  int      v4i;

typedef __attribute__((address_space(1))) v4i gv4i;   // global int4
typedef __attribute__((address_space(3))) v4i lv4i;   // LDS int4

static constexpr int BB   = 32;   // batch
static constexpr int CIN  = 32;
static constexpr int HH   = 192;
static constexpr int WW   = 192;
static constexpr int COUT = 32;
static constexpr int WTILE = 64;          // w positions per block
static constexpr int XCOLS = WTILE + 2;   // 66 (halo)
static constexpr int XPAD  = 40;          // halfs per (row,col) entry (80B, 16B-aligned)
static constexpr int ROWS  = 4;           // output rows per block
static constexpr int XROWS = ROWS + 2;    // staged input rows (halo)

#if __has_builtin(__builtin_amdgcn_global_load_async_to_lds_b128)
#define HAVE_ASYNC_LDS 1
#else
#define HAVE_ASYNC_LDS 0
#endif

// ---------------------------------------------------------------------------
// Kernel 1: fold the one-hot precision mask into the weights and convert to
// f16, laid out as wf[tap][cout][cin] so WMMA A-fragments are contiguous.
// ---------------------------------------------------------------------------
__global__ __launch_bounds__(256) void wconv_kernel(const float* __restrict__ weight,
                                                    const float* __restrict__ alpha_weight,
                                                    int n_bits,
                                                    _Float16* __restrict__ wf) {
    int total = 9 * COUT * CIN;
    for (int idx = blockIdx.x * blockDim.x + threadIdx.x; idx < total;
         idx += gridDim.x * blockDim.x) {
        int cin = idx % CIN;
        int co  = (idx / CIN) % COUT;
        int tap = idx / (CIN * COUT);
        int kh = tap / 3, kw = tap % 3;
        // argmax over the bit dimension (first index wins ties, as jnp.argmax)
        float best = alpha_weight[co];
        int   arg  = 0;
        for (int b = 1; b < n_bits; ++b) {
            float v = alpha_weight[b * COUT + co];
            if (v > best) { best = v; arg = b; }
        }
        float s  = (arg == 0) ? 1.0f : 0.0f;   // sw[0][co]
        float wv = weight[((co * CIN + cin) * 3 + kh) * 3 + kw] * s;
        wf[idx] = (_Float16)wv;
    }
}

// ---------------------------------------------------------------------------
// Kernel 2: implicit-GEMM 3x3 conv via v_wmma_f32_16x16x32_f16 + fused affine.
// Block = 256 threads = 8 waves; covers 4 output rows x 64 w x all 32 cout.
// Wave (wv): co-tile = wv&1 (16 couts), w-tile = wv>>1 (16 w).
// ---------------------------------------------------------------------------
__global__ __launch_bounds__(256) void conv_wmma_kernel(const float* __restrict__ x,
                                                        const _Float16* __restrict__ wf,
                                                        const float* __restrict__ alpha,
                                                        const float* __restrict__ b8,
                                                        const int* __restrict__ nb,
                                                        const int* __restrict__ nsh,
                                                        float* __restrict__ out) {
    __shared__ _Float16 xs[XROWS * XCOLS * XPAD];  // [row][col][cin(+pad)]  ~31.7 KB
    __shared__ _Float16 ws[9 * COUT * CIN];        // [tap][co][cin]         18 KB
    __shared__ float    affA[COUT];
    __shared__ float    affB[COUT];

    const int nWT = WW / WTILE;                    // 3
    const int nHB = HH / ROWS;                     // 48
    const int blk = blockIdx.x;
    const int wt  = blk % nWT;
    const int hb  = (blk / nWT) % nHB;
    const int b   = blk / (nWT * nHB);
    const int w0  = wt * WTILE;
    const int h0  = hb * ROWS;

    const int tid = threadIdx.x;

    // Per-channel affine: out = affA*acc + affB
    if (tid < COUT) {
        float sh = exp2f(-(float)nsh[tid]);
        affA[tid] = alpha[tid] * sh;
        affB[tid] = b8[tid] * exp2f((float)nb[tid]) * sh;
    }

    // Stage f16 weights into LDS. Pure byte copy -> async-to-LDS path (ASYNCcnt).
#if HAVE_ASYNC_LDS
    {
        const int n128 = 9 * COUT * CIN * (int)sizeof(_Float16) / 16;   // 1152 x 16B
        for (int i = tid; i < n128; i += 256) {
            __builtin_amdgcn_global_load_async_to_lds_b128(
                (gv4i*)((char*)wf + 16 * i),
                (lv4i*)((char*)ws + 16 * i),
                0, 0);
        }
#if __has_builtin(__builtin_amdgcn_s_wait_asynccnt)
        __builtin_amdgcn_s_wait_asynccnt(0);
#else
        asm volatile("s_wait_asynccnt 0" ::: "memory");
#endif
    }
#else
    {
        const uint32_t* src = (const uint32_t*)wf;
        uint32_t*       dst = (uint32_t*)ws;
        for (int i = tid; i < 9 * COUT * CIN / 2; i += 256) dst[i] = src[i];
    }
#endif

    // Stage x halo tile: rows h0-1..h0+4, cols w0-1..w0+64, all cin, zero-padded,
    // with f32 -> f16 conversion in flight (so no async path here).
    for (int i = tid; i < XROWS * CIN * XCOLS; i += 256) {
        int c   = i % XCOLS;                 // fastest -> coalesced global reads
        int cin = (i / XCOLS) % CIN;
        int r   = i / (XCOLS * CIN);
        int hr  = h0 + r - 1;
        int wc  = w0 + c - 1;
        float v = 0.0f;
        if (hr >= 0 && hr < HH && wc >= 0 && wc < WW)
            v = x[((b * CIN + cin) * HH + hr) * WW + wc];
        xs[(r * XCOLS + c) * XPAD + cin] = (_Float16)v;
    }
    __syncthreads();

    const int wv     = tid >> 5;
    const int lane   = tid & 31;
    const int hh     = lane >> 4;     // lane half (0/1)
    const int l16    = lane & 15;
    const int waveCo = wv & 1;        // cout tile
    const int waveW  = wv >> 1;       // w tile
    const int coA    = waveCo * 16 + l16;   // A-matrix row M = cout
    const int wN     = waveW * 16 + l16;    // B/D column N = w offset in block

    // Hoist all 9 A fragments (weights) into registers; reused for all 4 rows.
    // A layout (16x32 f16): elems 0..7 <- cin 8*hh.., elems 8..15 <- cin 16+8*hh..
    v16h afrag[9];
    #pragma unroll
    for (int t = 0; t < 9; ++t) {
        const _Float16* wrow = &ws[(t * COUT + coA) * CIN];
        v8h a0 = *(const v8h*)(wrow + 8 * hh);
        v8h a1 = *(const v8h*)(wrow + 16 + 8 * hh);
        afrag[t] = __builtin_shufflevector(a0, a1, 0, 1, 2, 3, 4, 5, 6, 7,
                                                   8, 9, 10, 11, 12, 13, 14, 15);
    }

    const int wOut = w0 + wN;
    #pragma unroll
    for (int r = 0; r < ROWS; ++r) {
        v8f acc = {};
        #pragma unroll
        for (int kh = 0; kh < 3; ++kh) {
            #pragma unroll
            for (int kw = 0; kw < 3; ++kw) {
                // B fragment (input 32x16): N = wN, elems 0..7 <- cin 8*hh.., elems 8..15 <- cin 16+8*hh..
                const _Float16* xrow =
                    &xs[((kh + r) * XCOLS + (wN + kw)) * XPAD];
                v8h b0 = *(const v8h*)(xrow + 8 * hh);
                v8h b1 = *(const v8h*)(xrow + 16 + 8 * hh);
                v16h bm = __builtin_shufflevector(b0, b1, 0, 1, 2, 3, 4, 5, 6, 7,
                                                          8, 9, 10, 11, 12, 13, 14, 15);
                acc = __builtin_amdgcn_wmma_f32_16x16x32_f16(
                    /*neg_a=*/false, afrag[kh * 3 + kw], /*neg_b=*/false, bm,
                    /*c_mod=*/(short)0, acc, /*reuse_a=*/false, /*reuse_b=*/false);
            }
        }
        // Epilogue: D row rr -> cout = waveCo*16 + rr + 8*hh, column N = wN.
        #pragma unroll
        for (int rr = 0; rr < 8; ++rr) {
            int   co = waveCo * 16 + rr + 8 * hh;
            float vr = acc[rr] * affA[co] + affB[co];
            out[((b * COUT + co) * HH + (h0 + r)) * WW + wOut] = vr;
        }
    }
}

// ---------------------------------------------------------------------------
extern "C" void kernel_launch(void* const* d_in, const int* in_sizes, int n_in,
                              void* d_out, int out_size, void* d_ws, size_t ws_size,
                              hipStream_t stream) {
    const float* x            = (const float*)d_in[0];
    const float* weight       = (const float*)d_in[1];
    const float* alpha_weight = (const float*)d_in[2];
    const float* alpha        = (const float*)d_in[3];
    const float* b8           = (const float*)d_in[4];
    const int*   nb           = (const int*)d_in[5];
    const int*   nsh          = (const int*)d_in[6];
    float*       out          = (float*)d_out;

    const int n_bits = in_sizes[2] / COUT;      // = 1 for BITS=[2]

    _Float16* wf = (_Float16*)d_ws;             // 9*32*32 f16 = 18432 B scratch

    wconv_kernel<<<36, 256, 0, stream>>>(weight, alpha_weight, n_bits, wf);

    const int nblocks = BB * (HH / ROWS) * (WW / WTILE); // 32*48*3 = 4608
    conv_wmma_kernel<<<nblocks, 256, 0, stream>>>(x, wf, alpha, b8, nb, nsh, out);
}
